// KAN_Convolution_3238405341764
// MI455X (gfx1250) — matrix operand order repaired and checked by reading, not compiled, not run on hardware
//
#include <hip/hip_runtime.h>
#include <hip/hip_bf16.h>

// ---------------- problem constants ----------------
#define IN_CH     32
#define OUT_CH    64
#define KERNEL    9
#define PAD       4
#define LEN       2048
#define BATCH     16
#define NCOEF     8              // GRID_SIZE + SPLINE_ORDER = 5+3
#define FEAT      (IN_CH * (NCOEF + 1))     // 288 features per position
#define KRED      (KERNEL * FEAT)           // 2592 GEMM reduction dim
#define LPAD      (LEN + 2 * PAD)           // 2056 padded positions
#define W_ELEMS   (OUT_CH * KRED)           // 165888 f16
#define A_ELEMS   ((size_t)BATCH * LPAD * FEAT)

typedef __attribute__((ext_vector_type(16))) _Float16 v16h;
typedef __attribute__((ext_vector_type(8)))  _Float16 v8h;
typedef __attribute__((ext_vector_type(8)))  float    v8f;
typedef __attribute__((ext_vector_type(4)))  float    v4f;

// ---------------------------------------------------------------------------
// Kernel 1: fuse base_w / spline_w / spline_scaler into one f16 weight matrix
//           W[o][k*288 + i*9 + c],  c=0 -> base, c=1..8 -> spline*scaler
// ---------------------------------------------------------------------------
__global__ __launch_bounds__(256) void kan_prep_w(
    const float* __restrict__ base_w,     // (64,32,9)
    const float* __restrict__ spline_w,   // (64,32,9,8)
    const float* __restrict__ scaler,     // (64,32,9)
    _Float16* __restrict__ W) {
  int tid = blockIdx.x * 256 + threadIdx.x;
  if (tid >= OUT_CH * IN_CH * KERNEL) return;
  int k = tid % KERNEL;
  int i = (tid / KERNEL) % IN_CH;
  int o = tid / (KERNEL * IN_CH);
  int src = (o * IN_CH + i) * KERNEL + k;
  float sc = scaler[src];
  _Float16* p = W + (size_t)o * KRED + k * FEAT + i * (NCOEF + 1);
  p[0] = (_Float16)base_w[src];
#pragma unroll
  for (int c = 0; c < NCOEF; ++c)
    p[1 + c] = (_Float16)(spline_w[(size_t)src * NCOEF + c] * sc);
}

// ---------------------------------------------------------------------------
// Kernel 2: per input element compute silu(x) and the 8 cubic B-spline bases.
//           Pad positions use x = 0 (silu(0)=0 but bs(0) != 0, as in reference).
//           Output: A[b][lp][i*9 + c]  (f16), lp in [0,2056)
// ---------------------------------------------------------------------------
__global__ __launch_bounds__(256) void kan_act(
    const float* __restrict__ x,          // (16,32,2048)
    _Float16* __restrict__ A) {
  int tid = blockIdx.x * 256 + threadIdx.x;
  if (tid >= BATCH * LPAD * IN_CH) return;
  int i  = tid & (IN_CH - 1);
  int lp = (tid / IN_CH) % LPAD;
  int b  = tid / (IN_CH * LPAD);

  float xv = 0.0f;
  int l = lp - PAD;
  if (l >= 0 && l < LEN) xv = x[((size_t)(b * IN_CH + i)) * LEN + l];

  // silu
  float s = xv / (1.0f + __expf(-xv));

  // Cox-de Boor, uniform knots t_j = -1 + (j-3)*0.4, j = 0..11
  const float h = 0.4f;
  float t[12];
#pragma unroll
  for (int j = 0; j < 12; ++j) t[j] = -1.0f + (float)(j - 3) * h;
  float bs[11];
#pragma unroll
  for (int j = 0; j < 11; ++j)
    bs[j] = (xv >= t[j] && xv < t[j + 1]) ? 1.0f : 0.0f;
#pragma unroll
  for (int k = 1; k <= 3; ++k) {
#pragma unroll
    for (int j = 0; j < 10; ++j) {
      if (j + k < 11) {
        float left  = (xv - t[j]) / (t[j + k] - t[j]);
        float right = (t[j + k + 1] - xv) / (t[j + k + 1] - t[j + 1]);
        bs[j] = left * bs[j] + right * bs[j + 1];
      }
    }
  }

  _Float16* p = A + (size_t)b * (LPAD * FEAT) + (size_t)lp * FEAT + i * (NCOEF + 1);
  p[0] = (_Float16)s;
#pragma unroll
  for (int c = 0; c < NCOEF; ++c) p[1 + c] = (_Float16)bs[c];
}

// ---------------------------------------------------------------------------
// Kernel 3: WMMA GEMM with LDS-staged weights.
//   out[b,o,l] = sum_r A_flat[b][l*288 + r] * W[o][r]
// Workgroup = 8 waves, all sharing one 32-wide o-half whose W rows
// (32 x 2592 f16 = 166 KB) are staged in LDS once; each wave then computes a
// 32(l) x 32(o) tile: per K-step 4x global_load_b128 (A, vmem pipe) +
// 4x ds_load_b128 (W, DS pipe) + 4x v_wmma_f32_16x16x32_f16.
// Fragment layouts per CDNA5 ISA tables:
//   A (16x32 f16):  lane = m + 16h holds K = {8h..8h+7, 16+8h..16+8h+7} of row m
//   B (32x16 f16):  lane = n + 16h holds K = {8h..8h+7, 16+8h..16+8h+7} of column n (= W row o)
//   D (16x16 f32):  lane = n + 16h, VGPR j holds D[8h + j][n]
// ---------------------------------------------------------------------------
static __device__ __forceinline__ v16h cat8(v8h lo, v8h hi) {
  return __builtin_shufflevector(lo, hi, 0, 1, 2, 3, 4, 5, 6, 7,
                                         8, 9, 10, 11, 12, 13, 14, 15);
}

__global__ __launch_bounds__(256) void kan_gemm(
    const _Float16* __restrict__ A,       // (16, 2056*288) f16
    const _Float16* __restrict__ W,       // (64, 2592) f16
    float* __restrict__ out) {            // (16, 64, 2048) f32
  __shared__ _Float16 Wlds[32 * KRED];    // 165,888 B (CDNA5: 320 KB/WGP)

  const int tid  = threadIdx.x;
  const int lane = tid & 31;
  const int wave = tid >> 5;
  const int ot2  = blockIdx.x & 1;        // which 32-wide o half (shared by WG)
  const int lg   = blockIdx.x >> 1;       // 0..127 : group of 8 l-tiles

  // ---- cooperative stage: this block's 32 W rows -> LDS (b128 copies) ----
  {
    const uint4* __restrict__ src = (const uint4*)(W + (size_t)ot2 * 32 * KRED);
    uint4* dst = (uint4*)Wlds;
    const int n128 = (32 * KRED) / 8;     // 10368 x 16B
    for (int idx = tid; idx < n128; idx += 256) dst[idx] = src[idx];
  }
  __syncthreads();

  const int gl0 = (lg * 8 + wave) << 5;   // global l among B*L
  const int b   = gl0 >> 11;              // / 2048
  const int l0  = gl0 & (LEN - 1);
  const int m   = lane & 15;
  const int h   = lane >> 4;

  const _Float16* Ab  = A + (size_t)b * (LPAD * FEAT);
  // two l-tiles; A per-lane halves at +8h / +16+8h
  const _Float16* ar0 = Ab + (size_t)(l0 + m) * FEAT + 8 * h;
  const _Float16* ar1 = ar0 + (size_t)16 * FEAT;
  // two o-tiles in LDS; B per-lane halves at +8h / +16+8h
  const _Float16* wl0 = Wlds + (size_t)m * KRED + 8 * h;
  const _Float16* wl1 = wl0 + (size_t)16 * KRED;

  v8f acc00 = {}, acc01 = {}, acc10 = {}, acc11 = {};

  for (int r = 0; r < KRED; r += 32) {
    // prefetch A a few K-steps ahead (speculative, streaming L2-resident rows)
    __builtin_prefetch((const void*)(ar0 + r + 128), 0, 0);
    __builtin_prefetch((const void*)(ar1 + r + 128), 0, 0);

    v8h a0l = *(const v8h*)(ar0 + r);
    v8h a0h = *(const v8h*)(ar0 + r + 16);
    v8h a1l = *(const v8h*)(ar1 + r);
    v8h a1h = *(const v8h*)(ar1 + r + 16);
    v8h b0l = *(const v8h*)(wl0 + r);
    v8h b0h = *(const v8h*)(wl0 + r + 16);
    v8h b1l = *(const v8h*)(wl1 + r);
    v8h b1h = *(const v8h*)(wl1 + r + 16);

    v16h a0 = cat8(a0l, a0h);
    v16h a1 = cat8(a1l, a1h);
    v16h w0 = cat8(b0l, b0h);
    v16h w1 = cat8(b1l, b1h);

    acc00 = __builtin_amdgcn_wmma_f32_16x16x32_f16(false, a0, false, w0,
                                                   (short)0, acc00, false, false);
    acc01 = __builtin_amdgcn_wmma_f32_16x16x32_f16(false, a0, false, w1,
                                                   (short)0, acc01, false, false);
    acc10 = __builtin_amdgcn_wmma_f32_16x16x32_f16(false, a1, false, w0,
                                                   (short)0, acc10, false, false);
    acc11 = __builtin_amdgcn_wmma_f32_16x16x32_f16(false, a1, false, w1,
                                                   (short)0, acc11, false, false);
  }

  // D epilogue: lane (n,h), VGPR j -> D[8h+j][n]; 8 consecutive l => 2x float4
#pragma unroll
  for (int lt = 0; lt < 2; ++lt) {
#pragma unroll
    for (int ot = 0; ot < 2; ++ot) {
      v8f acc = (lt == 0) ? (ot == 0 ? acc00 : acc01)
                          : (ot == 0 ? acc10 : acc11);
      int o = ot2 * 32 + ot * 16 + m;
      int l = l0 + lt * 16 + 8 * h;
      float* p = out + ((size_t)(b * OUT_CH + o) << 11) + l;
      *(v4f*)(p)     = __builtin_shufflevector(acc, acc, 0, 1, 2, 3);
      *(v4f*)(p + 4) = __builtin_shufflevector(acc, acc, 4, 5, 6, 7);
    }
  }
}

// ---------------------------------------------------------------------------
extern "C" void kernel_launch(void* const* d_in, const int* in_sizes, int n_in,
                              void* d_out, int out_size, void* d_ws, size_t ws_size,
                              hipStream_t stream) {
  (void)in_sizes; (void)n_in; (void)out_size; (void)ws_size;
  const float* x        = (const float*)d_in[0];
  const float* base_w   = (const float*)d_in[1];
  const float* spline_w = (const float*)d_in[2];
  const float* scaler   = (const float*)d_in[3];
  float* out = (float*)d_out;

  _Float16* W = (_Float16*)d_ws;                                // 331,776 B
  _Float16* A = (_Float16*)((char*)d_ws + (size_t)W_ELEMS * 2); // 18.9 MB

  int nw = OUT_CH * IN_CH * KERNEL;                 // 18432
  kan_prep_w<<<(nw + 255) / 256, 256, 0, stream>>>(base_w, spline_w, scaler, W);

  int na = BATCH * LPAD * IN_CH;                    // 1,052,672
  kan_act<<<(na + 255) / 256, 256, 0, stream>>>(x, A);

  // 256 blocks = 128 l-groups (8 x 32-l tiles) x 2 o-halves; 8 waves/block
  kan_gemm<<<256, 256, 0, stream>>>(A, W, out);
}
